// ReferenceMHA_59657095741551
// MI455X (gfx1250) — compile-verified, matrix operand order checked by reading, not run on hardware
//
#include <hip/hip_runtime.h>
#include <hip/hip_bf16.h>

// ---------------------------------------------------------------------------
// MHA forward for MI455X (gfx1250), bf16 WMMA + async global->LDS staging.
// Pipeline: cvt(fp32->bf16) -> QKV GEMMs -> flash attention -> out GEMM.
// ---------------------------------------------------------------------------

typedef __bf16 bf16_t;
typedef bf16_t bf16x16 __attribute__((ext_vector_type(16)));
typedef float  f32x8   __attribute__((ext_vector_type(8)));

union Frag { bf16x16 v; uint4 q[2]; };

#define DIMD   1024
#define SEQ    2048
#define BATCH  2
#define NHEAD  16
#define HDIM   64
#define MTOT   (BATCH * SEQ)   // 4096 rows

static __device__ __forceinline__ f32x8 wmma_bf16(bf16x16 a, bf16x16 b, f32x8 c) {
  return __builtin_amdgcn_wmma_f32_16x16x32_bf16(
      false, a, false, b, (short)0, c, false, false);
}

// LDS byte offset of a __shared__ object (addrspace(3) pointers are 32-bit).
static __device__ __forceinline__ unsigned lds_off(const void* p) {
  return (unsigned)(unsigned long long)
      (__attribute__((address_space(3))) const void*)p;
}

// Async DMA: 16 bytes per lane, global -> LDS, tracked by ASYNCcnt.
static __device__ __forceinline__ void async_ld16(unsigned lds, const void* g) {
  asm volatile("global_load_async_to_lds_b128 %0, %1, off"
               :: "v"(lds), "v"((unsigned long long)g)
               : "memory");
}
static __device__ __forceinline__ void wait_async0() {
  asm volatile("s_wait_asynccnt 0" ::: "memory");
}

// ------------------------------ fp32 -> bf16 -------------------------------
__global__ __launch_bounds__(256) void cvt_f32_bf16(
    const float* __restrict__ src, bf16_t* __restrict__ dst, int n4) {
  int i = blockIdx.x * blockDim.x + threadIdx.x;
  if (i < n4) {
    float4 f = ((const float4*)src)[i];
    union { bf16_t h[4]; ushort4 u; } o;
    o.h[0] = (bf16_t)f.x; o.h[1] = (bf16_t)f.y;
    o.h[2] = (bf16_t)f.z; o.h[3] = (bf16_t)f.w;
    ((ushort4*)dst)[i] = o.u;
  }
}

// ------------------------------ tiled GEMM ---------------------------------
// Y[m][n] = sum_k A[m][k] * W[n][k]   (A @ W^T), M=4096, N=K=1024.
// mode 0: bf16 [m][n];  mode 1: bf16 transposed -> Vt[b][h][hd][tok];
// mode 2: fp32 [m][n].
__global__ __launch_bounds__(256) void gemm128(
    const bf16_t* __restrict__ A, const bf16_t* __restrict__ W,
    void* __restrict__ out, int mode) {
  __shared__ bf16_t As[2][128][40];   // double-buffered, rows 16B-aligned
  __shared__ bf16_t Bs[2][128][40];

  const int tid  = threadIdx.x;
  const int lane = tid & 31, w = tid >> 5;
  const int r = lane & 15, hi = lane >> 4, kb8 = hi * 8;
  const int m0 = blockIdx.y * 128, n0 = blockIdx.x * 128;
  const int wm = (w & 3) * 32, wn = (w >> 2) * 64;   // 4x2 wave grid

  f32x8 acc[2][4] = {};

  // per-thread staging slots: 16 bf16 (half) of one row (two 16B async loads each)
  const int ldrow = tid >> 1, ldseg = (tid & 1) * 16;
  const bf16_t* ag = A + (size_t)(m0 + ldrow) * DIMD + ldseg;
  const bf16_t* wg = W + (size_t)(n0 + ldrow) * DIMD + ldseg;
  unsigned aoff[2], boff[2];
  aoff[0] = lds_off(&As[0][ldrow][ldseg]);
  aoff[1] = lds_off(&As[1][ldrow][ldseg]);
  boff[0] = lds_off(&Bs[0][ldrow][ldseg]);
  boff[1] = lds_off(&Bs[1][ldrow][ldseg]);

  // prefetch K-tile 0 into buffer 0
  async_ld16(aoff[0],      ag);
  async_ld16(aoff[0] + 16, ag + 8);
  async_ld16(boff[0],      wg);
  async_ld16(boff[0] + 16, wg + 8);

  for (int k0 = 0; k0 < DIMD; k0 += 32) {
    const int buf = (k0 >> 5) & 1;
    wait_async0();        // my async loads for this tile have landed
    __syncthreads();      // everyone's have

    if (k0 + 32 < DIMD) { // prefetch next tile into the other buffer
      const bf16_t* an = ag + k0 + 32;
      const bf16_t* wn2 = wg + k0 + 32;
      async_ld16(aoff[buf ^ 1],      an);
      async_ld16(aoff[buf ^ 1] + 16, an + 8);
      async_ld16(boff[buf ^ 1],      wn2);
      async_ld16(boff[buf ^ 1] + 16, wn2 + 8);
    }

    const bf16_t (*Asb)[40] = As[buf];
    const bf16_t (*Bsb)[40] = Bs[buf];
    Frag af[2], bfr[4];
#pragma unroll
    for (int i = 0; i < 2; i++) {
      af[i].q[0] = *(const uint4*)&Asb[wm + i * 16 + r][kb8];
      af[i].q[1] = *(const uint4*)&Asb[wm + i * 16 + r][kb8 + 16];
    }
#pragma unroll
    for (int j = 0; j < 4; j++) {
      bfr[j].q[0] = *(const uint4*)&Bsb[wn + j * 16 + r][kb8];
      bfr[j].q[1] = *(const uint4*)&Bsb[wn + j * 16 + r][kb8 + 16];
    }
#pragma unroll
    for (int i = 0; i < 2; i++)
#pragma unroll
      for (int j = 0; j < 4; j++)
        acc[i][j] = wmma_bf16(af[i].v, bfr[j].v, acc[i][j]);
  }

#pragma unroll
  for (int i = 0; i < 2; i++) {
#pragma unroll
    for (int j = 0; j < 4; j++) {
      const int col = n0 + wn + j * 16 + r;
#pragma unroll
      for (int rr = 0; rr < 8; rr++) {
        const int m = m0 + wm + i * 16 + hi * 8 + rr;
        const float v = acc[i][j][rr];
        if (mode == 2) {
          ((float*)out)[(size_t)m * DIMD + col] = v;
        } else if (mode == 0) {
          ((bf16_t*)out)[(size_t)m * DIMD + col] = (bf16_t)v;
        } else {
          const int b = m >> 11, tok = m & (SEQ - 1);
          const int h = col >> 6, hd = col & (HDIM - 1);
          ((bf16_t*)out)[(((size_t)(b * NHEAD + h)) * HDIM + hd) * SEQ + tok] =
              (bf16_t)v;
        }
      }
    }
  }
}

// --------------------------- flash attention -------------------------------
// Q,K layout: [b][tok][h][hd]; Vt layout: [b][h][hd][tok].
// One WG = 128 query rows of one (b,h); 8 waves x 16 rows; key blocks of 32.
__global__ __launch_bounds__(256) void attn_flash(
    const bf16_t* __restrict__ Q, const bf16_t* __restrict__ Kb,
    const bf16_t* __restrict__ Vt, bf16_t* __restrict__ O) {
  __shared__ bf16_t Ks[32][72];        // 32 keys x 64 d (+8 pad)
  __shared__ bf16_t Vts[64][40];       // 64 d x 32 keys (+8 pad)
  __shared__ bf16_t Ps[8][16][40];     // per-wave P transpose scratch

  const int tid = threadIdx.x, lane = tid & 31, w = tid >> 5;
  const int r = lane & 15, hi = lane >> 4, kb8 = hi * 8;
  const int nqb = SEQ / 128;
  const int bh = blockIdx.x / nqb, qb = blockIdx.x % nqb;
  const int b = bh >> 4, h = bh & (NHEAD - 1);
  const int q0 = qb * 128 + w * 16;

  // Q fragments (2 d-chunks of 32), loaded once from global
  Frag qa[2];
  {
    const bf16_t* qrow =
        Q + (((size_t)(b * SEQ + q0 + r)) * NHEAD + h) * HDIM;
#pragma unroll
    for (int c = 0; c < 2; c++) {
      qa[c].q[0] = *(const uint4*)(qrow + c * 32 + kb8);
      qa[c].q[1] = *(const uint4*)(qrow + c * 32 + kb8 + 16);
    }
  }

  // per-thread async staging slots (16B each)
  const int krow = tid >> 3, kseg = (tid & 7) * 8;      // K: 32 x 64
  const int vrow = tid >> 2, vseg = (tid & 3) * 8;      // Vt: 64 x 32
  const unsigned ksl = lds_off(&Ks[krow][kseg]);
  const unsigned vtl = lds_off(&Vts[vrow][vseg]);
  const bf16_t* kg =
      Kb + (((size_t)(b * SEQ + krow)) * NHEAD + h) * HDIM + kseg;
  const bf16_t* vg =
      Vt + (((size_t)(b * NHEAD + h)) * HDIM + vrow) * SEQ + vseg;

  f32x8 o[4] = {};
  float mrow[8], lrow[8];
#pragma unroll
  for (int rr = 0; rr < 8; rr++) { mrow[rr] = -1.0e30f; lrow[rr] = 0.0f; }

  const float SC = 0.125f * 1.44269504f;  // 1/sqrt(64) * log2(e)

  for (int j0 = 0; j0 < SEQ; j0 += 32) {
    __syncthreads();     // previous tile fully consumed by all waves
    async_ld16(ksl, kg + (size_t)j0 * NHEAD * HDIM);
    async_ld16(vtl, vg + j0);
    wait_async0();
    __syncthreads();

    // S = Q @ K^T for 2 key sub-tiles of 16
    f32x8 s[2] = {};
#pragma unroll
    for (int t = 0; t < 2; t++) {
#pragma unroll
      for (int c = 0; c < 2; c++) {
        Frag kf;
        kf.q[0] = *(const uint4*)&Ks[t * 16 + r][c * 32 + kb8];
        kf.q[1] = *(const uint4*)&Ks[t * 16 + r][c * 32 + kb8 + 16];
        s[t] = wmma_bf16(qa[c].v, kf.v, s[t]);
      }
    }

    // online softmax (log2 domain); write P (bf16) into per-wave LDS
#pragma unroll
    for (int rr = 0; rr < 8; rr++) {
      float v0 = s[0][rr] * SC, v1 = s[1][rr] * SC;
      float mx = fmaxf(v0, v1);
      mx = fmaxf(mx, __shfl_xor(mx, 1, 32));
      mx = fmaxf(mx, __shfl_xor(mx, 2, 32));
      mx = fmaxf(mx, __shfl_xor(mx, 4, 32));
      mx = fmaxf(mx, __shfl_xor(mx, 8, 32));
      float mnew = fmaxf(mrow[rr], mx);
      float fac = exp2f(mrow[rr] - mnew);
      float p0 = exp2f(v0 - mnew), p1 = exp2f(v1 - mnew);
      mrow[rr] = mnew;
      float ps = p0 + p1;
      ps += __shfl_xor(ps, 1, 32);
      ps += __shfl_xor(ps, 2, 32);
      ps += __shfl_xor(ps, 4, 32);
      ps += __shfl_xor(ps, 8, 32);
      lrow[rr] = lrow[rr] * fac + ps;
#pragma unroll
      for (int oc = 0; oc < 4; oc++) o[oc][rr] *= fac;
      const int prow = rr + hi * 8;
      Ps[w][prow][r]      = (bf16_t)p0;   // key col t=0
      Ps[w][prow][16 + r] = (bf16_t)p1;   // key col t=1
    }

    // DS ops of one wave are in-order; wait for the P stores, block reorder.
    asm volatile("s_wait_dscnt 0" ::: "memory");

    // O += P(16x32) @ Vblk(32x64)
    Frag pf;
    pf.q[0] = *(const uint4*)&Ps[w][r][kb8];
    pf.q[1] = *(const uint4*)&Ps[w][r][kb8 + 16];
#pragma unroll
    for (int oc = 0; oc < 4; oc++) {
      Frag vb;
      vb.q[0] = *(const uint4*)&Vts[oc * 16 + r][kb8];
      vb.q[1] = *(const uint4*)&Vts[oc * 16 + r][kb8 + 16];
      o[oc] = wmma_bf16(pf.v, vb.v, o[oc]);
    }
  }

  // normalize and store O -> [b][tok][h][hd] (bf16)
#pragma unroll
  for (int rr = 0; rr < 8; rr++) {
    const float inv = 1.0f / lrow[rr];
    const int tok = q0 + rr + hi * 8;
    const size_t base = (((size_t)(b * SEQ + tok)) * NHEAD + h) * HDIM;
#pragma unroll
    for (int oc = 0; oc < 4; oc++)
      O[base + oc * 16 + r] = (bf16_t)(o[oc][rr] * inv);
  }
}

// ------------------------------- launcher ----------------------------------
extern "C" void kernel_launch(void* const* d_in, const int* in_sizes, int n_in,
                              void* d_out, int out_size, void* d_ws,
                              size_t ws_size, hipStream_t stream) {
  const float* x  = (const float*)d_in[0];
  const float* wq = (const float*)d_in[1];
  const float* wk = (const float*)d_in[2];
  const float* wv = (const float*)d_in[3];
  const float* wo = (const float*)d_in[4];

  const size_t XE = (size_t)MTOT * DIMD;       // 4M elems
  const size_t WE = (size_t)DIMD * DIMD;       // 1M elems

  bf16_t* ws  = (bf16_t*)d_ws;
  bf16_t* xb  = ws;
  bf16_t* wqb = xb + XE;
  bf16_t* wkb = wqb + WE;
  bf16_t* wvb = wkb + WE;
  bf16_t* wob = wvb + WE;
  bf16_t* qb  = wob + WE;
  bf16_t* kb  = qb + XE;
  bf16_t* vtb = kb + XE;
  bf16_t* ob  = vtb + XE;

  // 1) fp32 -> bf16
  cvt_f32_bf16<<<dim3((unsigned)(XE / 4 / 256)), dim3(256), 0, stream>>>(x, xb, (int)(XE / 4));
  cvt_f32_bf16<<<dim3((unsigned)(WE / 4 / 256)), dim3(256), 0, stream>>>(wq, wqb, (int)(WE / 4));
  cvt_f32_bf16<<<dim3((unsigned)(WE / 4 / 256)), dim3(256), 0, stream>>>(wk, wkb, (int)(WE / 4));
  cvt_f32_bf16<<<dim3((unsigned)(WE / 4 / 256)), dim3(256), 0, stream>>>(wv, wvb, (int)(WE / 4));
  cvt_f32_bf16<<<dim3((unsigned)(WE / 4 / 256)), dim3(256), 0, stream>>>(wo, wob, (int)(WE / 4));

  // 2) projections (M=4096, N=1024): Q, K normal layout; V transposed per head
  dim3 ggrid(DIMD / 128, MTOT / 128);
  gemm128<<<ggrid, dim3(256), 0, stream>>>(xb, wqb, (void*)qb, 0);
  gemm128<<<ggrid, dim3(256), 0, stream>>>(xb, wkb, (void*)kb, 0);
  gemm128<<<ggrid, dim3(256), 0, stream>>>(xb, wvb, (void*)vtb, 1);

  // 3) flash attention: B*H*(SEQ/128) workgroups
  attn_flash<<<dim3(BATCH * NHEAD * (SEQ / 128)), dim3(256), 0, stream>>>(
      qb, kb, vtb, ob);

  // 4) output projection -> fp32 d_out
  gemm128<<<ggrid, dim3(256), 0, stream>>>(ob, wob, d_out, 2);
}
